// MHADecoder_67044439490702
// MI455X (gfx1250) — compile-verified, matrix operand order checked by reading, not run on hardware
//
#include <hip/hip_runtime.h>

#define Bsz  16
#define Nn   1024
#define Gg   1024
#define Hh   16
#define Dd   32
#define HDim 512
#define Ee   512

typedef __bf16 bf16;
typedef __attribute__((ext_vector_type(16))) __bf16 v16bf;
typedef __attribute__((ext_vector_type(8)))  __bf16 v8bf;
typedef __attribute__((ext_vector_type(8)))  float  v8f;

static __device__ __forceinline__ v8f wmma_bf16(v16bf a, v16bf b, v8f c) {
  // D = A(16x32) * B(32x16) + C, f32 accumulate
  return __builtin_amdgcn_wmma_f32_16x16x32_bf16(false, a, false, b, (short)0, c,
                                                 false, false);
}

// Fragment loader from a row-major tile [(>=16) x (>=32)], element (row,k).
// Per CDNA5 layout, lane half selects K {0..7,16..23} vs {8..15,24..31}: two
// contiguous 16-byte runs -> two b128 loads. Requires 16B-aligned rows.
static __device__ __forceinline__ v16bf load_frag_rm(const bf16* t, int ld, int lane) {
  int m = lane & 15, hf = lane >> 4;
  const bf16* p = t + m * ld + hf * 8;
  v8bf lo = *(const v8bf*)p;
  v8bf hi = *(const v8bf*)(p + 16);
  return __builtin_shufflevector(lo, hi, 0, 1, 2, 3, 4, 5, 6, 7,
                                 8, 9, 10, 11, 12, 13, 14, 15);
}

static __device__ __forceinline__ v8bf cvt8(const float* __restrict__ p) {
  v8bf v;
#pragma unroll
  for (int i = 0; i < 8; ++i) v[i] = (bf16)p[i];
  return v;
}

// ---------------------------------------------------------------- mean over N
__global__ __launch_bounds__(256) void mean_kernel(const float* __restrict__ nodes,
                                                   float* __restrict__ gm) {
  int b = blockIdx.x;
  for (int e = threadIdx.x; e < Ee; e += 256) {
    float s = 0.f;
    for (int n = 0; n < Nn; ++n) s += nodes[((size_t)b * Nn + n) * Ee + e];
    gm[b * Ee + e] = s * (1.0f / Nn);
  }
}

// --------------------------------------------------- f32 -> bf16 (8 per thread)
__global__ __launch_bounds__(256) void cvt_kernel(const float* __restrict__ src,
                                                  bf16* __restrict__ dst, int n8) {
  int i = blockIdx.x * 256 + threadIdx.x;
  if (i < n8) *(v8bf*)(dst + (size_t)i * 8) = cvt8(src + (size_t)i * 8);
}

// --------------------- f32 [K x N] -> bf16 transposed [N x K] (LDS 32x32 tiles)
__global__ __launch_bounds__(256) void cvt_t_kernel(const float* __restrict__ src,
                                                    bf16* __restrict__ dst) {
  __shared__ float tile[32][33];
  int bx = blockIdx.x % (HDim / 32);  // col tile of src
  int by = blockIdx.x / (HDim / 32);  // row tile of src
  int tx = threadIdx.x & 31, ty = threadIdx.x >> 5;
  for (int i = ty; i < 32; i += 8)
    tile[i][tx] = src[(size_t)(by * 32 + i) * HDim + bx * 32 + tx];
  __syncthreads();
  for (int i = ty; i < 32; i += 8)
    dst[(size_t)(bx * 32 + i) * Ee + by * 32 + tx] = (bf16)tile[tx][i];
}

// ------------------------------------------- graph query: [B,E] @ [E,HD] (tiny)
__global__ __launch_bounds__(256) void qg_kernel(const float* __restrict__ gm,
                                                 const float* __restrict__ Wqg,
                                                 float* __restrict__ Qg) {
  int id = blockIdx.x * 256 + threadIdx.x;
  if (id >= Bsz * HDim) return;
  int b = id / HDim, j = id % HDim;
  float s = 0.f;
  for (int e = 0; e < Ee; ++e) s += gm[b * Ee + e] * Wqg[e * HDim + j];
  Qg[id] = s;
}

// --------------- bf16 GEMM: C[M,N] = A[M,K] @ Bt[N,K]^T (+bias), bf16 output
// 256 thr = 8 waves, 64x64 block tile, waves 4(row) x 2(col of 32)
template <bool HAS_BIAS>
__global__ __launch_bounds__(256) void gemm_kernel(const bf16* __restrict__ A,
                                                   const bf16* __restrict__ Bt,
                                                   const float* __restrict__ bias,
                                                   bf16* __restrict__ C,
                                                   int M, int N, int K) {
  __shared__ bf16 As[64][40];
  __shared__ bf16 Bs[64][40];  // n-major
  int tid = threadIdx.x, wave = tid >> 5, lane = tid & 31;
  int wr = wave & 3, wc = wave >> 2;
  int row0 = blockIdx.x * 64, col0 = blockIdx.y * 64;
  int sr = (tid * 8) >> 5, scol = (tid * 8) & 31;  // staging: 8 contiguous/thread
  v8f c0 = {}, c1 = {};
  for (int k0 = 0; k0 < K; k0 += 32) {
    *(v8bf*)&As[sr][scol] = *(const v8bf*)&A[(size_t)(row0 + sr) * K + k0 + scol];
    *(v8bf*)&Bs[sr][scol] = *(const v8bf*)&Bt[(size_t)(col0 + sr) * K + k0 + scol];
    __syncthreads();
    v16bf a  = load_frag_rm(&As[wr * 16][0], 40, lane);
    v16bf b0 = load_frag_rm(&Bs[wc * 32][0], 40, lane);
    v16bf b1 = load_frag_rm(&Bs[wc * 32 + 16][0], 40, lane);
    c0 = wmma_bf16(a, b0, c0);
    c1 = wmma_bf16(a, b1, c1);
    __syncthreads();
  }
  int n = lane & 15, hf = lane >> 4;
#pragma unroll
  for (int r = 0; r < 8; ++r) {
    int m = r + 8 * hf;
    int gr = row0 + wr * 16 + m;
    int gc = col0 + wc * 32 + n;
    float v0 = c0[r], v1 = c1[r];
    if (HAS_BIAS) { v0 += bias[gc]; v1 += bias[gc + 16]; }
    C[(size_t)gr * N + gc]      = (bf16)v0;
    C[(size_t)gr * N + gc + 16] = (bf16)v1;
  }
}

// ---- fused Q projection: Q = first@Wqf + last@Wql + broadcast(Qg[b]) (Wq* pre-T)
__global__ __launch_bounds__(256) void qproj_kernel(const float* __restrict__ first,
                                                    const float* __restrict__ last,
                                                    const bf16* __restrict__ WqfT,
                                                    const bf16* __restrict__ WqlT,
                                                    const float* __restrict__ Qg,
                                                    bf16* __restrict__ Q) {
  __shared__ bf16 A1[64][40], A2[64][40];
  __shared__ bf16 B1s[64][40], B2s[64][40];  // n-major
  int tid = threadIdx.x, wave = tid >> 5, lane = tid & 31;
  int wr = wave & 3, wc = wave >> 2;
  int row0 = blockIdx.x * 64, col0 = blockIdx.y * 64;
  int sr = (tid * 8) >> 5, scol = (tid * 8) & 31;
  v8f c0 = {}, c1 = {};
  for (int k0 = 0; k0 < Ee; k0 += 32) {
    size_t aoff = (size_t)(row0 + sr) * Ee + k0 + scol;
    *(v8bf*)&A1[sr][scol] = cvt8(first + aoff);
    *(v8bf*)&A2[sr][scol] = cvt8(last + aoff);
    size_t boff = (size_t)(col0 + sr) * Ee + k0 + scol;
    *(v8bf*)&B1s[sr][scol] = *(const v8bf*)(WqfT + boff);
    *(v8bf*)&B2s[sr][scol] = *(const v8bf*)(WqlT + boff);
    __syncthreads();
    v16bf a1  = load_frag_rm(&A1[wr * 16][0], 40, lane);
    v16bf a2  = load_frag_rm(&A2[wr * 16][0], 40, lane);
    v16bf b1a = load_frag_rm(&B1s[wc * 32][0], 40, lane);
    v16bf b1b = load_frag_rm(&B1s[wc * 32 + 16][0], 40, lane);
    v16bf b2a = load_frag_rm(&B2s[wc * 32][0], 40, lane);
    v16bf b2b = load_frag_rm(&B2s[wc * 32 + 16][0], 40, lane);
    c0 = wmma_bf16(a1, b1a, c0);
    c0 = wmma_bf16(a2, b2a, c0);
    c1 = wmma_bf16(a1, b1b, c1);
    c1 = wmma_bf16(a2, b2b, c1);
    __syncthreads();
  }
  int n = lane & 15, hf = lane >> 4;
  int b = row0 / Gg;  // 64-row tiles never straddle a batch (G % 64 == 0)
#pragma unroll
  for (int r = 0; r < 8; ++r) {
    int m = r + 8 * hf;
    int gr = row0 + wr * 16 + m;
    int gc = col0 + wc * 32 + n;
    Q[(size_t)gr * HDim + gc]      = (bf16)(c0[r] + Qg[b * HDim + gc]);
    Q[(size_t)gr * HDim + gc + 16] = (bf16)(c1[r] + Qg[b * HDim + gc + 16]);
  }
}

// ------------------- flash attention: per wave 16 q-rows, online softmax, P@V
__global__ __launch_bounds__(256) void attn_kernel(const bf16* __restrict__ Q,
                                                   const bf16* __restrict__ Kp,
                                                   const bf16* __restrict__ Vp,
                                                   const float* __restrict__ mask,
                                                   bf16* __restrict__ Oc) {
  int bid = blockIdx.x;                 // B * H * (G/128)
  int gt = bid % (Gg / 128);
  int h  = (bid / (Gg / 128)) % Hh;
  int b  = bid / ((Gg / 128) * Hh);
  int tid = threadIdx.x, wave = tid >> 5, lane = tid & 31;
  int g0 = gt * 128 + wave * 16;        // this wave's 16 query rows

  __shared__ bf16 Kt[32][40];           // [n][d] row-major
  __shared__ bf16 Vtt[32][40];          // [d][n] (transposed for B-frag reads)
  __shared__ bf16 Pt[8][16][40];        // per-wave P bounce (C-layout -> A-layout)

  v16bf qa = load_frag_rm(Q + ((size_t)b * Gg + g0) * HDim + h * Dd, HDim, lane);

  v8f o0 = {}, o1 = {};
  float mrow[8], lrow[8];
#pragma unroll
  for (int r = 0; r < 8; ++r) { mrow[r] = -1e30f; lrow[r] = 0.f; }

  const float scale = 0.17677669529663687f;  // 1/sqrt(D)
  int nIdx = lane & 15, hf = lane >> 4;

  for (int n0 = 0; n0 < Nn; n0 += 32) {
    if (tid < 128) {                    // K: vector load + vector LDS store
      int idx = tid * 8, r = idx >> 5, c = idx & 31;
      *(v8bf*)&Kt[r][c] =
          *(const v8bf*)&Kp[((size_t)b * Nn + n0 + r) * HDim + h * Dd + c];
    } else {                            // V: vector load, transposed LDS store
      int idx = (tid - 128) * 8, r = idx >> 5, c = idx & 31;
      v8bf vv = *(const v8bf*)&Vp[((size_t)b * Nn + n0 + r) * HDim + h * Dd + c];
#pragma unroll
      for (int i = 0; i < 8; ++i) Vtt[c + i][r] = vv[i];
    }
    __syncthreads();
    // scores: S[m][n] = sum_d Q[m][d]*K[n][d] (B-frag == row-major read of Kt)
    v16bf kb0 = load_frag_rm(&Kt[0][0], 40, lane);
    v16bf kb1 = load_frag_rm(&Kt[16][0], 40, lane);
    v8f s0 = {}, s1 = {};
    s0 = wmma_bf16(qa, kb0, s0);
    s1 = wmma_bf16(qa, kb1, s1);
#pragma unroll
    for (int r = 0; r < 8; ++r) {
      int m = r + 8 * hf;
      size_t mb = ((size_t)b * Gg + g0 + m) * Nn + n0;
      float v0 = s0[r] * scale + mask[mb + nIdx];
      float v1 = s1[r] * scale + mask[mb + 16 + nIdx];
      float mx = fmaxf(v0, v1);
      for (int off = 8; off >= 1; off >>= 1) mx = fmaxf(mx, __shfl_xor(mx, off, 32));
      float mnew = fmaxf(mrow[r], mx);
      float corr = __expf(mrow[r] - mnew);
      float p0 = __expf(v0 - mnew);
      float p1 = __expf(v1 - mnew);
      float ps = p0 + p1;
      for (int off = 8; off >= 1; off >>= 1) ps += __shfl_xor(ps, off, 32);
      lrow[r] = lrow[r] * corr + ps;
      mrow[r] = mnew;
      o0[r] *= corr;
      o1[r] *= corr;
      Pt[wave][m][nIdx]      = (bf16)p0;
      Pt[wave][m][nIdx + 16] = (bf16)p1;
    }
    // O += P(16x32) @ V(32x32)
    v16bf pa  = load_frag_rm(&Pt[wave][0][0], 40, lane);
    v16bf vb0 = load_frag_rm(&Vtt[0][0], 40, lane);   // cols d 0..15
    v16bf vb1 = load_frag_rm(&Vtt[16][0], 40, lane);  // cols d 16..31
    o0 = wmma_bf16(pa, vb0, o0);
    o1 = wmma_bf16(pa, vb1, o1);
    __syncthreads();
  }
#pragma unroll
  for (int r = 0; r < 8; ++r) {
    int m = r + 8 * hf;
    float inv = 1.0f / lrow[r];
    size_t base = ((size_t)b * Gg + g0 + m) * HDim + h * Dd;
    Oc[base + nIdx]      = (bf16)(o0[r] * inv);
    Oc[base + 16 + nIdx] = (bf16)(o1[r] * inv);
  }
}

// --------- pointer head: score2 = mh @ nodes^T, 10*tanh + mask, softmax -> out
__global__ __launch_bounds__(256) void pointer_kernel(const bf16* __restrict__ MH,
                                                      const bf16* __restrict__ nodes,
                                                      const float* __restrict__ mask,
                                                      float* __restrict__ out) {
  int bid = blockIdx.x;                 // B * (G/16)
  int gt = bid % (Gg / 16);
  int b  = bid / (Gg / 16);
  int g0 = gt * 16;
  int tid = threadIdx.x, wave = tid >> 5, lane = tid & 31;
  __shared__ float sc[16][Nn];          // 64 KB score tile

  const float iscale = 0.044194173824159216f;  // 1/sqrt(E)
  int nIdx = lane & 15, hf = lane >> 4;
  for (int ci = 0; ci < Nn / (16 * 8); ++ci) {
    int n0 = (wave + 8 * ci) * 16;
    v8f c = {};
    for (int e0 = 0; e0 < Ee; e0 += 32) {
      v16bf a  = load_frag_rm(MH + ((size_t)b * Gg + g0) * Ee + e0, Ee, lane);
      // B[k=e][n] = nodes[b][n0+n][e0+k]: row-major read of nodes is B^T
      v16bf bf = load_frag_rm(nodes + ((size_t)b * Nn + n0) * Ee + e0, Ee, lane);
      c = wmma_bf16(a, bf, c);
    }
#pragma unroll
    for (int r = 0; r < 8; ++r) {
      int m = r + 8 * hf;
      sc[m][n0 + nIdx] = 10.0f * tanhf(c[r] * iscale) +
                         mask[((size_t)b * Gg + g0 + m) * Nn + n0 + nIdx];
    }
  }
  __syncthreads();
  // per-wave row softmax: wave w handles rows w and w+8
  for (int rr = 0; rr < 2; ++rr) {
    int row = wave + rr * 8;
    float mx = -1e30f;
    for (int j = lane; j < Nn; j += 32) mx = fmaxf(mx, sc[row][j]);
    for (int off = 16; off >= 1; off >>= 1) mx = fmaxf(mx, __shfl_xor(mx, off, 32));
    float sum = 0.f;
    for (int j = lane; j < Nn; j += 32) {
      float e = __expf(sc[row][j] - mx);
      sc[row][j] = e;
      sum += e;
    }
    for (int off = 16; off >= 1; off >>= 1) sum += __shfl_xor(sum, off, 32);
    float inv = 1.0f / sum;
    for (int j = lane; j < Nn; j += 32)
      out[((size_t)b * Gg + g0 + row) * Nn + j] = sc[row][j] * inv;
  }
}

// ---------------------------------------------------------------------------
extern "C" void kernel_launch(void* const* d_in, const int* in_sizes, int n_in,
                              void* d_out, int out_size, void* d_ws, size_t ws_size,
                              hipStream_t stream) {
  const float* nodes = (const float*)d_in[0];
  const float* first = (const float*)d_in[1];
  const float* last  = (const float*)d_in[2];
  const float* mask  = (const float*)d_in[3];
  const float* Wqg   = (const float*)d_in[4];
  const float* Wqf   = (const float*)d_in[5];
  const float* Wql   = (const float*)d_in[6];
  const float* Wk    = (const float*)d_in[7];
  const float* Wv    = (const float*)d_in[8];
  const float* Wc    = (const float*)d_in[9];
  const float* bc    = (const float*)d_in[10];
  float* out = (float*)d_out;

  char* p = (char*)d_ws;
  auto take = [&](size_t bytes) {
    char* r = p;
    p += (bytes + 255) & ~(size_t)255;
    return r;
  };
  float* gm     = (float*)take((size_t)Bsz * Ee * 4);
  float* Qg     = (float*)take((size_t)Bsz * HDim * 4);
  bf16* nodesB  = (bf16*)take((size_t)Bsz * Nn * Ee * 2);
  bf16* WkT     = (bf16*)take((size_t)Ee * HDim * 2);
  bf16* WvT     = (bf16*)take((size_t)Ee * HDim * 2);
  bf16* WqfT    = (bf16*)take((size_t)Ee * HDim * 2);
  bf16* WqlT    = (bf16*)take((size_t)Ee * HDim * 2);
  bf16* WcT     = (bf16*)take((size_t)HDim * Ee * 2);
  bf16* Kp      = (bf16*)take((size_t)Bsz * Nn * HDim * 2);
  bf16* Vp      = (bf16*)take((size_t)Bsz * Nn * HDim * 2);
  bf16* Qp      = (bf16*)take((size_t)Bsz * Gg * HDim * 2);
  bf16* Oc      = (bf16*)take((size_t)Bsz * Gg * HDim * 2);
  bf16* MHb     = (bf16*)take((size_t)Bsz * Gg * Ee * 2);

  mean_kernel<<<Bsz, 256, 0, stream>>>(nodes, gm);

  int nNodes8 = Bsz * Nn * Ee / 8;
  cvt_kernel<<<(nNodes8 + 255) / 256, 256, 0, stream>>>(nodes, nodesB, nNodes8);

  int tgrid = (HDim / 32) * (Ee / 32);
  cvt_t_kernel<<<tgrid, 256, 0, stream>>>(Wk, WkT);
  cvt_t_kernel<<<tgrid, 256, 0, stream>>>(Wv, WvT);
  cvt_t_kernel<<<tgrid, 256, 0, stream>>>(Wqf, WqfT);
  cvt_t_kernel<<<tgrid, 256, 0, stream>>>(Wql, WqlT);
  cvt_t_kernel<<<tgrid, 256, 0, stream>>>(Wc, WcT);

  qg_kernel<<<(Bsz * HDim + 255) / 256, 256, 0, stream>>>(gm, Wqg, Qg);

  dim3 gkv(Bsz * Nn / 64, HDim / 64);
  gemm_kernel<false><<<gkv, 256, 0, stream>>>(nodesB, WkT, nullptr, Kp,
                                              Bsz * Nn, HDim, Ee);
  gemm_kernel<false><<<gkv, 256, 0, stream>>>(nodesB, WvT, nullptr, Vp,
                                              Bsz * Nn, HDim, Ee);

  qproj_kernel<<<dim3(Bsz * Gg / 64, HDim / 64), 256, 0, stream>>>(first, last, WqfT,
                                                                   WqlT, Qg, Qp);

  attn_kernel<<<Bsz * Hh * (Gg / 128), 256, 0, stream>>>(Qp, Kp, Vp, mask, Oc);

  gemm_kernel<true><<<dim3(Bsz * Gg / 64, Ee / 64), 256, 0, stream>>>(
      Oc, WcT, bc, MHb, Bsz * Gg, Ee, HDim);

  pointer_kernel<<<Bsz * (Gg / 16), 256, 0, stream>>>(MHb, nodesB, mask, out);
}
